// AttentionPairBias_84731114815935
// MI455X (gfx1250) — matrix operand order repaired and checked by reading, not compiled, not run on hardware
//
#include <hip/hip_runtime.h>
#include <stdint.h>

// ---------------------------------------------------------------------------
// AttentionPairBias for MI455X (gfx1250, wave32, WMMA bf16 16x16x32)
// B=1, L=1024, D=512, P=128, H=16, dh=32
// ---------------------------------------------------------------------------

static constexpr int LSEQ = 1024;
static constexpr int DMODEL = 512;
static constexpr int PDIM = 128;
static constexpr int NH = 16;
static constexpr int DH = 32;

typedef __attribute__((ext_vector_type(16))) __bf16 v16bf;
typedef __attribute__((ext_vector_type(8)))  float  v8f;
typedef __attribute__((ext_vector_type(4)))  float  fx4;
typedef __attribute__((ext_vector_type(4)))  unsigned int ux4;

union Frag {
  ux4 u[2];
  unsigned short s[16];
  v16bf v;
};

__device__ __forceinline__ unsigned short f2bf(float f) {
  union { float f; unsigned int u; } cv; cv.f = f;
  unsigned int u = cv.u;
  u += 0x7fffu + ((u >> 16) & 1u);       // round-to-nearest-even
  return (unsigned short)(u >> 16);
}

__device__ __forceinline__ v8f wmma_bf16(v16bf a, v16bf b, v8f c) {
  return __builtin_amdgcn_wmma_f32_16x16x32_bf16(false, a, false, b,
                                                 (short)0, c, false, false);
}

__device__ __forceinline__ v8f zero_v8f() {
  v8f c;
#pragma unroll
  for (int i = 0; i < 8; ++i) c[i] = 0.0f;
  return c;
}

__device__ __forceinline__ fx4 nt_load_f4(const float* p) {
  return __builtin_nontemporal_load((const fx4*)p);
}

// Async copy of 32 contiguous bytes per lane: global -> LDS (CDNA5 async path,
// tracked by ASYNCcnt). lds_off is the raw LDS byte offset for this lane.
__device__ __forceinline__ void async_copy32(unsigned lds_off, const void* gaddr) {
  asm volatile("global_load_async_to_lds_b128 %0, %1, off\n\t"
               "global_load_async_to_lds_b128 %0, %1, off offset:16"
               :: "v"(lds_off), "v"((unsigned long long)(uintptr_t)gaddr)
               : "memory");
}

__device__ __forceinline__ void wait_asynccnt_le8() {
  asm volatile("s_wait_asynccnt 0x8" ::: "memory");
}

// Load A fragment (16x32 bf16) from rowptr = A + row*lda + kblock.
__device__ __forceinline__ v16bf load_a_frag(const unsigned short* rowptr, int hl) {
  Frag f;
  f.u[0] = *(const ux4*)(rowptr + 16 * hl);
  f.u[1] = *(const ux4*)(rowptr + 16 * hl + 8);
  return f.v;
}

// ---------------------------------------------------------------------------
// Pack a row-major f32 weight W[K][N] into bf16 WMMA B-fragments.
// tile t = kblk * (N/16) + nblk ; packed[(t*32 + lane)*16 + j]
// ---------------------------------------------------------------------------
__global__ void pack_b_kernel(const float* __restrict__ W,
                              unsigned short* __restrict__ out,
                              int K, int N) {
  int lane = threadIdx.x;
  int t = blockIdx.x;
  int nT = N >> 4;
  int kb = t / nT, nt = t % nT;
  int n = nt * 16 + (lane & 15);
  int k0 = kb * 32 + 16 * (lane >> 4);
  unsigned short* o = out + ((size_t)t * 32 + lane) * 16;
#pragma unroll
  for (int j = 0; j < 16; ++j)
    o[j] = f2bf(W[(size_t)(k0 + j) * N + n]);
}

// ---------------------------------------------------------------------------
// LayerNorm over D=512 per row of single; output bf16. One block per row.
// ---------------------------------------------------------------------------
__global__ void ln_single_kernel(const float* __restrict__ x,
                                 const float* __restrict__ g,
                                 const float* __restrict__ b,
                                 unsigned short* __restrict__ out) {
  __shared__ float red[2][4];
  int row = blockIdx.x;
  int t = threadIdx.x;
  float vals[4];
  float s = 0.f, q = 0.f;
#pragma unroll
  for (int i = 0; i < 4; ++i) {
    float f = x[(size_t)row * DMODEL + t + i * 128];
    vals[i] = f; s += f; q += f * f;
  }
#pragma unroll
  for (int o = 16; o > 0; o >>= 1) {
    s += __shfl_xor(s, o, 32);
    q += __shfl_xor(q, o, 32);
  }
  if ((t & 31) == 0) { red[0][t >> 5] = s; red[1][t >> 5] = q; }
  __syncthreads();
  s = red[0][0] + red[0][1] + red[0][2] + red[0][3];
  q = red[1][0] + red[1][1] + red[1][2] + red[1][3];
  float mean = s / (float)DMODEL;
  float var = q / (float)DMODEL - mean * mean;
  float rstd = rsqrtf(var + 1e-5f);
#pragma unroll
  for (int i = 0; i < 4; ++i) {
    int idx = t + i * 128;
    out[(size_t)row * DMODEL + idx] =
        f2bf((vals[i] - mean) * rstd * g[idx] + b[idx]);
  }
}

// ---------------------------------------------------------------------------
// Generic WMMA GEMM: C[M,N] = A[M,K](bf16) * Bpacked(bf16)
//  mode 0: bf16 out                mode 1: (acc + bias[n]) * scale -> bf16
//  mode 2: sigmoid(acc) -> f32     mode 3: acc -> f32
//  mode 4: bf16 out TRANSPOSED (out[n*M + m]), for V
// ---------------------------------------------------------------------------
__global__ void gemm_wmma_kernel(const unsigned short* __restrict__ A,
                                 const unsigned short* __restrict__ Bp,
                                 int M, int N, int K,
                                 const float* __restrict__ bias,
                                 float scale, int mode,
                                 unsigned short* __restrict__ outb,
                                 float* __restrict__ outf) {
  int lane = threadIdx.x & 31;
  int wv = threadIdx.x >> 5;
  int nT = N >> 4;
  int tile = blockIdx.x * 4 + wv;
  if (tile >= (M >> 4) * nT) return;
  int mt = tile / nT, nt = tile % nT;
  int hl = lane >> 4;
  int ln15 = lane & 15;
  int row = mt * 16 + ln15;

  v8f c = zero_v8f();
  const unsigned short* arow = A + (size_t)row * K;
  const unsigned short* bp = Bp + ((size_t)nt * 32 + lane) * 16;
  const size_t bstep = (size_t)nT * 512;   // one k-block of packed tiles
  for (int kb = 0; kb < K; kb += 32) {
    v16bf a = load_a_frag(arow + kb, hl);
    Frag bf;
    bf.u[0] = *(const ux4*)(bp);
    bf.u[1] = *(const ux4*)(bp + 8);
    bp += bstep;
    c = wmma_bf16(a, bf.v, c);
  }

  int n = nt * 16 + ln15;
#pragma unroll
  for (int r = 0; r < 8; ++r) {
    int m = mt * 16 + r + 8 * hl;
    float acc = c[r];
    if (mode == 1) acc = (acc + bias[n]) * scale;
    if (mode == 2) {
      outf[(size_t)m * N + n] = 1.0f / (1.0f + __expf(-acc));
    } else if (mode == 3) {
      outf[(size_t)m * N + n] = acc;
    } else if (mode == 4) {
      outb[(size_t)n * M + m] = f2bf(acc);
    } else {
      outb[(size_t)m * N + n] = f2bf(acc);
    }
  }
}

// ---------------------------------------------------------------------------
// Fused pair LayerNorm (over P=128) + projection by Wb (128x16) via WMMA.
// Block = 512 threads = 16 waves, covering a 16i x 16j super-tile.
// Wave wv handles i = i0+wv (16 j-rows): NT-load 8KB of pair, LN, 4x WMMA.
// D tiles staged through a 16KB LDS cube [h][j][i], then written out fully
// coalesced to bias2[h][j][i].
// ---------------------------------------------------------------------------
__global__ void pair_bias_kernel(const float* __restrict__ pair,
                                 const float* __restrict__ g,
                                 const float* __restrict__ b,
                                 const unsigned short* __restrict__ wbp,
                                 float* __restrict__ bias2) {
  __shared__ float T[16 * 16 * 16];   // [h][j][i]
  int tid = threadIdx.x;
  int lane = tid & 31, wv = tid >> 5;  // wv = local i (16 waves)
  int hl = lane >> 4;
  int ln15 = lane & 15;
  const int nST = (LSEQ / 16) * (LSEQ / 16);   // 4096 super-tiles
  const int stride = gridDim.x;

  // Preload Wb fragments (K=128 -> 4 k-blocks, single N tile)
  v16bf wb[4];
#pragma unroll
  for (int kb = 0; kb < 4; ++kb) {
    Frag f;
    const unsigned short* p = wbp + ((size_t)kb * 32 + lane) * 16;
    f.u[0] = *(const ux4*)(p);
    f.u[1] = *(const ux4*)(p + 8);
    wb[kb] = f.v;
  }

  // Preload gamma/beta at this lane's K offsets (64 values each)
  float gv[64], bv[64];
#pragma unroll
  for (int kb = 0; kb < 4; ++kb) {
#pragma unroll
    for (int e = 0; e < 16; ++e) {
      int k = kb * 32 + 16 * hl + e;
      gv[kb * 16 + e] = g[k];
      bv[kb * 16 + e] = b[k];
    }
  }

  for (int st = blockIdx.x; st < nST; st += stride) {
    int i0 = (st >> 6) << 4;
    int j0 = (st & 63) << 4;
    const float* rp =
        pair + ((size_t)(i0 + wv) * LSEQ + j0 + ln15) * PDIM;

    // Prefetch next super-tile's rows (global_prefetch_b8)
    int stn = st + stride;
    if (stn < nST) {
      const float* np = pair +
          ((size_t)(((stn >> 6) << 4) + wv) * LSEQ + ((stn & 63) << 4) + ln15) * PDIM +
          hl * 64;
      __builtin_prefetch(np, 0, 0);
      __builtin_prefetch(np + 32, 0, 0);
    }

    float vals[64];
    float sum = 0.f, sq = 0.f;
#pragma unroll
    for (int kb = 0; kb < 4; ++kb) {
#pragma unroll
      for (int ch = 0; ch < 2; ++ch) {
        int off = kb * 32 + 16 * hl + ch * 8;
        fx4 f0 = nt_load_f4(rp + off);
        fx4 f1 = nt_load_f4(rp + off + 4);
        int vi = kb * 16 + ch * 8;
#pragma unroll
        for (int e = 0; e < 4; ++e) { vals[vi + e] = f0[e]; vals[vi + 4 + e] = f1[e]; }
#pragma unroll
        for (int e = 0; e < 4; ++e) { sum += f0[e] + f1[e]; sq += f0[e]*f0[e] + f1[e]*f1[e]; }
      }
    }
    sum += __shfl_xor(sum, 16, 32);
    sq  += __shfl_xor(sq, 16, 32);
    float mean = sum / (float)PDIM;
    float var = sq / (float)PDIM - mean * mean;
    float rstd = rsqrtf(var + 1e-5f);

    v8f c = zero_v8f();
#pragma unroll
    for (int kb = 0; kb < 4; ++kb) {
      Frag a;
#pragma unroll
      for (int e = 0; e < 16; ++e)
        a.s[e] = f2bf((vals[kb * 16 + e] - mean) * rstd * gv[kb * 16 + e] +
                      bv[kb * 16 + e]);
      c = wmma_bf16(a.v, wb[kb], c);
    }

    // D tile: rows = j (r + 8*hl), cols = h (ln15); stage into LDS cube.
#pragma unroll
    for (int r = 0; r < 8; ++r)
      T[(ln15 * 16 + r + 8 * hl) * 16 + wv] = c[r];
    __syncthreads();

    // Coalesced write-out: each thread 8 consecutive i for one (h,j).
    {
      int e = tid * 8;
      int h = e >> 8, jj = (e >> 4) & 15, ii = e & 15;
      float* dst = bias2 + ((size_t)h * LSEQ + j0 + jj) * LSEQ + i0 + ii;
      const fx4* srcp = (const fx4*)&T[e];
      fx4 a0 = srcp[0];
      fx4 a1 = srcp[1];
      *(fx4*)dst = a0;
      *(fx4*)(dst + 4) = a1;
    }
    __syncthreads();
  }
}

// ---------------------------------------------------------------------------
// Flash attention per (head, 16-query tile): QK^T WMMA + bias + mask,
// streaming softmax, P via LDS re-fragmentation, P*V WMMA, gate, /sum.
// K and V operands are double-buffered through LDS with
// global_load_async_to_lds_b128 (ASYNCcnt), hiding L2 latency behind WMMA.
// bias2 is [h][j][i]; vt is V transposed [d][key].
// ---------------------------------------------------------------------------
__global__ void attention_kernel(const unsigned short* __restrict__ qbf,
                                 const unsigned short* __restrict__ kbf,
                                 const unsigned short* __restrict__ vt,
                                 const float* __restrict__ bias2,
                                 const unsigned char* __restrict__ mask,
                                 const float* __restrict__ gate,
                                 unsigned short* __restrict__ outb) {
  __shared__ unsigned short plds[4][16 * 32];
  // [wave][buf][chunk: Kb0,Kb1,Vb0,Vb1][lane][16 bf16]  = 32 KB
  __shared__ unsigned short stage[4][2][4][32][16];

  int lane = threadIdx.x & 31;
  int wv = threadIdx.x >> 5;
  int tile = blockIdx.x * 4 + wv;     // 1024 tiles, exact
  int head = tile >> 6;
  int i0 = (tile & 63) << 4;
  int hl = lane >> 4;
  int ln15 = lane & 15;

  // Q A-fragment: rows i0..i0+15, cols head*32..+31 (dh = one K step)
  const unsigned short* qrow =
      qbf + (size_t)(i0 + ln15) * DMODEL + head * DH;
  v16bf qa = load_a_frag(qrow, hl);

  // Issue async K/V staging for key-tile jt into buffer buf.
  auto issue = [&](int jt, int buf) {
    const unsigned short* kr0 =
        kbf + (size_t)(jt + ln15) * DMODEL + head * DH + 16 * hl;
    const unsigned short* kr1 = kr0 + 16 * DMODEL;
    const unsigned short* v0 =
        vt + (size_t)(head * DH + ln15) * LSEQ + jt + 16 * hl;
    const unsigned short* v1 = v0 + (size_t)16 * LSEQ;
    unsigned base = (unsigned)(uintptr_t)&stage[wv][buf][0][lane][0];
    async_copy32(base, kr0);
    async_copy32(base + 1024, kr1);
    async_copy32(base + 2048, v0);
    async_copy32(base + 3072, v1);
  };

  float rmax[8], rsum[8];
  v8f o0 = zero_v8f(), o1 = zero_v8f();
#pragma unroll
  for (int r = 0; r < 8; ++r) { rmax[r] = -1e30f; rsum[r] = 0.f; }

  issue(0, 0);

  for (int jt = 0; jt < LSEQ; jt += 32) {
    int buf = (jt >> 5) & 1;
    issue((jt + 32) & (LSEQ - 1), buf ^ 1);   // prefetch next tile

    bool mk0 = mask[jt + ln15] != 0;
    bool mk1 = mask[jt + 16 + ln15] != 0;

    // Bias: lane j fixed, i contiguous -> float4 x2 per score sub-tile (NT)
    const float* bp0 =
        bias2 + ((size_t)head * LSEQ + jt + ln15) * LSEQ + i0 + 8 * hl;
    const float* bp1 = bp0 + (size_t)16 * LSEQ;
    fx4 x0 = nt_load_f4(bp0), x1 = nt_load_f4(bp0 + 4);
    fx4 y0 = nt_load_f4(bp1), y1 = nt_load_f4(bp1 + 4);
    float bv0[8], bv1[8];
#pragma unroll
    for (int e = 0; e < 4; ++e) {
      bv0[e] = x0[e]; bv0[4 + e] = x1[e];
      bv1[e] = y0[e]; bv1[4 + e] = y1[e];
    }

    // Wait for this buffer's async loads (only the 8 just-issued may remain)
    wait_asynccnt_le8();
    const ux4* sp = (const ux4*)&stage[wv][buf][0][lane][0];
    Frag b0, b1, vb0, vb1;
    b0.u[0]  = sp[0];   b0.u[1]  = sp[1];
    b1.u[0]  = sp[64];  b1.u[1]  = sp[65];
    vb0.u[0] = sp[128]; vb0.u[1] = sp[129];
    vb1.u[0] = sp[192]; vb1.u[1] = sp[193];

    v8f s0 = zero_v8f(), s1 = zero_v8f();
    s0 = wmma_bf16(qa, b0.v, s0);
    s1 = wmma_bf16(qa, b1.v, s1);

    float sc0[8], sc1[8];
#pragma unroll
    for (int r = 0; r < 8; ++r) {
      sc0[r] = mk0 ? s0[r] + bv0[r] : -1e9f;
      sc1[r] = mk1 ? s1[r] + bv1[r] : -1e9f;
    }

#pragma unroll
    for (int r = 0; r < 8; ++r) {
      float mx = fmaxf(sc0[r], sc1[r]);
#pragma unroll
      for (int o = 1; o < 16; o <<= 1) mx = fmaxf(mx, __shfl_xor(mx, o, 16));
      float nm = fmaxf(rmax[r], mx);
      float fac = __expf(rmax[r] - nm);
      rmax[r] = nm;
      float p0 = __expf(sc0[r] - nm);
      float p1 = __expf(sc1[r] - nm);
      float ps = p0 + p1;
#pragma unroll
      for (int o = 1; o < 16; o <<= 1) ps += __shfl_xor(ps, o, 16);
      rsum[r] = rsum[r] * fac + ps;
      o0[r] *= fac;
      o1[r] *= fac;
      plds[wv][(r + 8 * hl) * 32 + ln15] = f2bf(p0);
      plds[wv][(r + 8 * hl) * 32 + 16 + ln15] = f2bf(p1);
    }
    __syncthreads();

    // P A-fragment from LDS
    const unsigned short* pr = &plds[wv][ln15 * 32];
    Frag pa;
    pa.u[0] = *(const ux4*)(pr + 16 * hl);
    pa.u[1] = *(const ux4*)(pr + 16 * hl + 8);
    __syncthreads();

    o0 = wmma_bf16(pa.v, vb0.v, o0);
    o1 = wmma_bf16(pa.v, vb1.v, o1);
  }

#pragma unroll
  for (int r = 0; r < 8; ++r) {
    int mrow = i0 + r + 8 * hl;
    float gn = gate[(size_t)mrow * NH + head];
    float inv = 1.0f / rsum[r];
    unsigned short* orow = outb + (size_t)mrow * DMODEL + head * DH;
    orow[ln15]      = f2bf(o0[r] * inv * gn);
    orow[16 + ln15] = f2bf(o1[r] * inv * gn);
  }
}

// ---------------------------------------------------------------------------
// Host-side launch
// ---------------------------------------------------------------------------
extern "C" void kernel_launch(void* const* d_in, const int* in_sizes, int n_in,
                              void* d_out, int out_size, void* d_ws, size_t ws_size,
                              hipStream_t stream) {
  (void)in_sizes; (void)n_in; (void)out_size; (void)ws_size;

  const float* single = (const float*)d_in[0];
  const float* pair   = (const float*)d_in[1];
  const unsigned char* mask = (const unsigned char*)d_in[2];
  const float* ln_s_g = (const float*)d_in[3];
  const float* ln_s_b = (const float*)d_in[4];
  const float* Wq     = (const float*)d_in[5];
  const float* bq     = (const float*)d_in[6];
  const float* Wk     = (const float*)d_in[7];
  const float* Wv     = (const float*)d_in[8];
  const float* ln_p_g = (const float*)d_in[9];
  const float* ln_p_b = (const float*)d_in[10];
  const float* Wb     = (const float*)d_in[11];
  const float* Wg     = (const float*)d_in[12];
  const float* Wo     = (const float*)d_in[13];
  float* out = (float*)d_out;

  char* ws = (char*)d_ws;
  size_t off = 0;
  auto alloc = [&](size_t bytes) -> size_t {
    size_t o = off;
    off = (off + bytes + 255) & ~(size_t)255;
    return o;
  };

  const size_t DD_BF = (size_t)DMODEL * DMODEL * sizeof(unsigned short);
  unsigned short* wq_p = (unsigned short*)(ws + alloc(DD_BF));
  unsigned short* wk_p = (unsigned short*)(ws + alloc(DD_BF));
  unsigned short* wv_p = (unsigned short*)(ws + alloc(DD_BF));
  unsigned short* wo_p = (unsigned short*)(ws + alloc(DD_BF));
  unsigned short* wg_p = (unsigned short*)(ws + alloc((size_t)DMODEL * NH * 2));
  unsigned short* wb_p = (unsigned short*)(ws + alloc((size_t)PDIM * NH * 2));
  unsigned short* sbf  = (unsigned short*)(ws + alloc((size_t)LSEQ * DMODEL * 2));
  unsigned short* qbf  = (unsigned short*)(ws + alloc((size_t)LSEQ * DMODEL * 2));
  unsigned short* kbf  = (unsigned short*)(ws + alloc((size_t)LSEQ * DMODEL * 2));
  unsigned short* vt   = (unsigned short*)(ws + alloc((size_t)LSEQ * DMODEL * 2));
  float* gatef         = (float*)(ws + alloc((size_t)LSEQ * NH * 4));
  unsigned short* attout = (unsigned short*)(ws + alloc((size_t)LSEQ * DMODEL * 2));
  float* bias2         = (float*)(ws + alloc((size_t)NH * LSEQ * LSEQ * 4));

  // 1) Pack weights into WMMA B-fragment order (bf16)
  hipLaunchKernelGGL(pack_b_kernel, dim3((DMODEL / 32) * (DMODEL / 16)), dim3(32), 0, stream, Wq, wq_p, DMODEL, DMODEL);
  hipLaunchKernelGGL(pack_b_kernel, dim3((DMODEL / 32) * (DMODEL / 16)), dim3(32), 0, stream, Wk, wk_p, DMODEL, DMODEL);
  hipLaunchKernelGGL(pack_b_kernel, dim3((DMODEL / 32) * (DMODEL / 16)), dim3(32), 0, stream, Wv, wv_p, DMODEL, DMODEL);
  hipLaunchKernelGGL(pack_b_kernel, dim3((DMODEL / 32) * (DMODEL / 16)), dim3(32), 0, stream, Wo, wo_p, DMODEL, DMODEL);
  hipLaunchKernelGGL(pack_b_kernel, dim3((DMODEL / 32) * (NH / 16)), dim3(32), 0, stream, Wg, wg_p, DMODEL, NH);
  hipLaunchKernelGGL(pack_b_kernel, dim3((PDIM / 32) * (NH / 16)), dim3(32), 0, stream, Wb, wb_p, PDIM, NH);

  // 2) LayerNorm single -> bf16
  hipLaunchKernelGGL(ln_single_kernel, dim3(LSEQ), dim3(128), 0, stream, single, ln_s_g, ln_s_b, sbf);

  // 3) Q/K/V and gate projections
  const float qscale = 0.17677669529663689f;  // 1/sqrt(32)
  int tiles_qkv = (LSEQ / 16) * (DMODEL / 16);   // 2048
  hipLaunchKernelGGL(gemm_wmma_kernel, dim3(tiles_qkv / 4), dim3(128), 0, stream,
                     sbf, wq_p, LSEQ, DMODEL, DMODEL, bq, qscale, 1, qbf, (float*)nullptr);
  hipLaunchKernelGGL(gemm_wmma_kernel, dim3(tiles_qkv / 4), dim3(128), 0, stream,
                     sbf, wk_p, LSEQ, DMODEL, DMODEL, (const float*)nullptr, 1.0f, 0, kbf, (float*)nullptr);
  hipLaunchKernelGGL(gemm_wmma_kernel, dim3(tiles_qkv / 4), dim3(128), 0, stream,
                     sbf, wv_p, LSEQ, DMODEL, DMODEL, (const float*)nullptr, 1.0f, 4, vt, (float*)nullptr);
  int tiles_gate = (LSEQ / 16) * (NH / 16);      // 64
  hipLaunchKernelGGL(gemm_wmma_kernel, dim3(tiles_gate / 4), dim3(128), 0, stream,
                     sbf, wg_p, LSEQ, NH, DMODEL, (const float*)nullptr, 1.0f, 2,
                     (unsigned short*)nullptr, gatef);

  // 4) Fused pair LN + Wb projection -> bias2[h][j][i] (one pass over 512 MB)
  hipLaunchKernelGGL(pair_bias_kernel, dim3(1024), dim3(512), 0, stream,
                     pair, ln_p_g, ln_p_b, wb_p, bias2);

  // 5) Flash attention with bias + mask + gate (async LDS double buffering)
  hipLaunchKernelGGL(attention_kernel, dim3((NH * (LSEQ / 16)) / 4), dim3(128), 0, stream,
                     qbf, kbf, vt, bias2, mask, gatef, attout);

  // 6) Output projection -> f32 d_out
  hipLaunchKernelGGL(gemm_wmma_kernel, dim3(tiles_qkv / 4), dim3(128), 0, stream,
                     attout, wo_p, LSEQ, DMODEL, DMODEL, (const float*)nullptr, 1.0f, 3,
                     (unsigned short*)nullptr, out);
}
